// CausalAttention_38706245272094
// MI455X (gfx1250) — matrix operand order repaired and checked by reading, not compiled, not run on hardware
//
#include <hip/hip_runtime.h>

// ---------------- types ----------------
typedef __attribute__((ext_vector_type(8)))  float  v8f;
typedef __attribute__((ext_vector_type(16))) __bf16 v16bf;
typedef __attribute__((ext_vector_type(8)))  __bf16 v8bf;

union AF { v16bf v; v8bf h[2]; };

__device__ __forceinline__ __bf16 f2bf(float f) {
  union { float f; unsigned u; } a; a.f = f;
  unsigned r = a.u + 0x7FFFu + ((a.u >> 16) & 1u);   // round-to-nearest-even
  union { unsigned short s; __bf16 b; } c; c.s = (unsigned short)(r >> 16);
  return c.b;
}

#define BATCH 4
#define SEQ   2048
#define DIMD  1024
#define HEADS 16
#define DHEAD 64

// ---------------- LayerNorm + cast to bf16 ----------------
__global__ __launch_bounds__(256) void ln_kernel(const float* __restrict__ x,
                                                 const float* __restrict__ g,
                                                 const float* __restrict__ bta,
                                                 __bf16* __restrict__ xn) {
  __shared__ float red[256];
  const int row = blockIdx.x, tid = threadIdx.x;
  const float* xr = x + (size_t)row * DIMD;
  float v[4];
  float s = 0.f;
#pragma unroll
  for (int i = 0; i < 4; ++i) { v[i] = xr[tid + i * 256]; s += v[i]; }
  red[tid] = s; __syncthreads();
  for (int o = 128; o > 0; o >>= 1) { if (tid < o) red[tid] += red[tid + o]; __syncthreads(); }
  const float mu = red[0] * (1.f / DIMD);
  __syncthreads();
  s = 0.f;
#pragma unroll
  for (int i = 0; i < 4; ++i) { float d = v[i] - mu; s += d * d; }
  red[tid] = s; __syncthreads();
  for (int o = 128; o > 0; o >>= 1) { if (tid < o) red[tid] += red[tid + o]; __syncthreads(); }
  const float rstd = rsqrtf(red[0] * (1.f / DIMD) + 1e-5f);
  __bf16* outr = xn + (size_t)row * DIMD;
#pragma unroll
  for (int i = 0; i < 4; ++i) {
    int c = tid + i * 256;
    outr[c] = f2bf((v[i] - mu) * rstd * g[c] + bta[c]);
  }
}

// ---------------- fp32 -> bf16 cast ----------------
__global__ __launch_bounds__(256) void cast_kernel(const float* __restrict__ s,
                                                   __bf16* __restrict__ d, int n) {
  for (int i = blockIdx.x * 256 + threadIdx.x; i < n; i += gridDim.x * 256)
    d[i] = f2bf(s[i]);
}

// ---------------- WMMA GEMM: C[MxN] = A[MxK]*B[KxN], bf16 in, f32 acc --------
// Block tile 128x128, 8 waves of 32x64, K-step 32, double-buffered LDS with
// CDNA5 async copy (A tile) + prefetch two steps ahead.
template<int N, bool QKV>
__global__ __launch_bounds__(256) void gemm_kernel(const __bf16* __restrict__ A,
                                                   const __bf16* __restrict__ B,
                                                   __bf16* __restrict__ Oq,
                                                   float* __restrict__ Of,
                                                   int K) {
  __shared__ __bf16 lsA[2][128 * 40];   // [row][k], padded
  __shared__ __bf16 lsB[2][128 * 40];   // [col][k], transposed, padded
  const int tid = threadIdx.x;
  const int lane = tid & 31, w = tid >> 5;
  const int wm = w & 3, wn = w >> 2;
  const int lr = lane & 15, half = lane >> 4;
  const int m0 = blockIdx.y * 128;
  const int n0 = blockIdx.x * 128;

  const int ar = tid >> 1, ako = (tid & 1) * 16;   // A coop: 128 rows x 32 k
  const int br = tid >> 3, bco = (tid & 7) * 16;   // B coop: 32 k x 128 cols

  v8f acc[2][4] = {};

  // A tile: async copy global -> LDS (no VGPR round trip), 2 x b128 per lane
  auto issueA = [&](int k0, int buf) {
    const __bf16* src = A + (size_t)(m0 + ar) * K + k0 + ako;
    unsigned d0 = (unsigned)(size_t)(&lsA[buf][ar * 40 + ako]);
    unsigned long long s0 = (unsigned long long)(size_t)src;
    asm volatile("global_load_async_to_lds_b128 %0, %1, off"
                 :: "v"(d0), "v"(s0) : "memory");
    unsigned d1 = d0 + 16u;
    unsigned long long s1 = s0 + 16ull;
    asm volatile("global_load_async_to_lds_b128 %0, %1, off"
                 :: "v"(d1), "v"(s1) : "memory");
  };
  // B tile: coalesced read + transposed store (needed for B-fragment layout)
  auto issueB = [&](int k0, int buf) {
    const __bf16* src = B + (size_t)(k0 + br) * N + n0 + bco;
    v8bf b0 = *(const v8bf*)src;
    v8bf b1 = *(const v8bf*)(src + 8);
#pragma unroll
    for (int i = 0; i < 8; ++i) lsB[buf][(bco + i) * 40 + br] = b0[i];
#pragma unroll
    for (int i = 0; i < 8; ++i) lsB[buf][(bco + 8 + i) * 40 + br] = b1[i];
  };

  issueA(0, 0);
  issueB(0, 0);
  asm volatile("s_wait_asynccnt 0" ::: "memory");
  __syncthreads();

  const int NK = K >> 5;
  for (int kk = 0; kk < NK; ++kk) {
    const int p = kk & 1;

    AF af[2], bfr[4];
#pragma unroll
    for (int i = 0; i < 2; ++i) {
      int rr = wm * 32 + i * 16 + lr;
      af[i].h[0] = *(const v8bf*)&lsA[p][rr * 40 + half * 8];
      af[i].h[1] = *(const v8bf*)&lsA[p][rr * 40 + 16 + half * 8];
    }
#pragma unroll
    for (int j = 0; j < 4; ++j) {
      int cc = wn * 64 + j * 16 + lr;
      bfr[j].h[0] = *(const v8bf*)&lsB[p][cc * 40 + half * 8];
      bfr[j].h[1] = *(const v8bf*)&lsB[p][cc * 40 + 16 + half * 8];
    }

    if (kk + 1 < NK) {
      if (kk + 2 < NK) { // warm L2/WGP$ two tiles ahead
        __builtin_prefetch(A + (size_t)(m0 + ar) * K + (kk + 2) * 32 + ako, 0, 0);
        __builtin_prefetch(B + (size_t)((kk + 2) * 32 + br) * N + n0 + bco, 0, 0);
      }
      issueA((kk + 1) * 32, 1 - p);
      issueB((kk + 1) * 32, 1 - p);
    }

#pragma unroll
    for (int i = 0; i < 2; ++i)
#pragma unroll
      for (int j = 0; j < 4; ++j)
        acc[i][j] = __builtin_amdgcn_wmma_f32_16x16x32_bf16(
            false, af[i].v, false, bfr[j].v, (short)0, acc[i][j], false, false);

    asm volatile("s_wait_asynccnt 0" ::: "memory");
    __syncthreads();
  }

  // store (C layout: vgpr e -> row e + half*8, col = lr)
#pragma unroll
  for (int i = 0; i < 2; ++i)
#pragma unroll
    for (int j = 0; j < 4; ++j) {
      int gm = m0 + wm * 32 + i * 16 + half * 8;
      int gn = n0 + wn * 64 + j * 16 + lr;
#pragma unroll
      for (int e = 0; e < 8; ++e) {
        int row = gm + e;
        float val = acc[i][j][e];
        if (QKV) {
          int which = gn >> 10, cc = gn & 1023, hh = cc >> 6, dh = cc & 63;
          int bi = row >> 11, ni = row & 2047;
          size_t idx = ((((size_t)which * BATCH + bi) * HEADS + hh) * SEQ + ni) * DHEAD + dh;
          Oq[idx] = f2bf(val);
        } else {
          Of[(size_t)row * N + gn] = val;
        }
      }
    }
}

// ---------------- Flash-style causal attention ----------------
// grid = (BATCH*HEADS) * (SEQ/128); 8 waves, each owns 16 q-rows.
__global__ __launch_bounds__(256) void attn_kernel(const __bf16* __restrict__ Q,
                                                   const __bf16* __restrict__ Kk,
                                                   const __bf16* __restrict__ V,
                                                   __bf16* __restrict__ Aout) {
  __shared__ __bf16 vt[64 * 40];        // V transposed: [dh][k], padded
  __shared__ __bf16 plds[8 * 16 * 40];  // per-wave P staging
  const int tid = threadIdx.x, lane = tid & 31, w = tid >> 5;
  const int lr = lane & 15, half = lane >> 4;
  const int qb = blockIdx.x & 15;
  const int bh = blockIdx.x >> 4;
  const int hi = bh & 15, bi = bh >> 4;
  const size_t base = (size_t)bh * SEQ * DHEAD;
  const int q0 = qb * 128 + w * 16;

  AF qa[2]; // Q 16x64 A-fragments, kept in registers for the whole loop
  {
    const __bf16* qp = Q + base + (size_t)(q0 + lr) * DHEAD;
#pragma unroll
    for (int f = 0; f < 2; ++f) {
      qa[f].h[0] = *(const v8bf*)(qp + f * 32 + half * 8);
      qa[f].h[1] = *(const v8bf*)(qp + f * 32 + 16 + half * 8);
    }
  }

  float m_i[8], l_i[8];
#pragma unroll
  for (int e = 0; e < 8; ++e) { m_i[e] = -1e30f; l_i[e] = 0.f; }
  v8f o[4] = {};

  const int nkb = (qb + 1) * 4;
  __bf16* pbuf = plds + w * 16 * 40;

  for (int kb2 = 0; kb2 < nkb; ++kb2) {
    const int kc0 = kb2 * 32;
    { // cooperative V(32x64) -> vt[64][k] transpose
      int r = tid >> 3, co = (tid & 7) * 8;
      const __bf16* vp = V + base + (size_t)(kc0 + r) * DHEAD + co;
      v8bf vv = *(const v8bf*)vp;
#pragma unroll
      for (int i = 0; i < 8; ++i) vt[(co + i) * 40 + r] = vv[i];
    }
    __syncthreads();

    if (kc0 <= q0 + 15) {
      v8f s[2] = {};
#pragma unroll
      for (int t2 = 0; t2 < 2; ++t2) {
        const __bf16* kp = Kk + base + (size_t)(kc0 + t2 * 16 + lr) * DHEAD;
        AF b0, b1; // K rows loaded in the same k order as the Q A-fragments
        b0.h[0] = *(const v8bf*)(kp + half * 8);
        b0.h[1] = *(const v8bf*)(kp + 16 + half * 8);
        b1.h[0] = *(const v8bf*)(kp + 32 + half * 8);
        b1.h[1] = *(const v8bf*)(kp + 32 + 16 + half * 8);
        s[t2] = __builtin_amdgcn_wmma_f32_16x16x32_bf16(
            false, qa[0].v, false, b0.v, (short)0, s[t2], false, false);
        s[t2] = __builtin_amdgcn_wmma_f32_16x16x32_bf16(
            false, qa[1].v, false, b1.v, (short)0, s[t2], false, false);
      }

      float sv[2][8], fac[8];
#pragma unroll
      for (int t2 = 0; t2 < 2; ++t2)
#pragma unroll
        for (int e = 0; e < 8; ++e) sv[t2][e] = s[t2][e];

#pragma unroll
      for (int e = 0; e < 8; ++e) { // scale + causal mask
        int row = q0 + half * 8 + e;
#pragma unroll
        for (int t2 = 0; t2 < 2; ++t2) {
          int col = kc0 + t2 * 16 + lr;
          sv[t2][e] = (col <= row) ? sv[t2][e] * 0.125f : -1e30f;
        }
      }
#pragma unroll
      for (int e = 0; e < 8; ++e) { // online softmax (16-lane row reductions)
        float mx = fmaxf(sv[0][e], sv[1][e]);
        for (int m = 1; m < 16; m <<= 1) mx = fmaxf(mx, __shfl_xor(mx, m, 32));
        float nm = fmaxf(m_i[e], mx);
        float f = __expf(m_i[e] - nm);
        sv[0][e] = __expf(sv[0][e] - nm);
        sv[1][e] = __expf(sv[1][e] - nm);
        float rs = sv[0][e] + sv[1][e];
        for (int m = 1; m < 16; m <<= 1) rs += __shfl_xor(rs, m, 32);
        l_i[e] = l_i[e] * f + rs;
        m_i[e] = nm;
        fac[e] = f;
      }
#pragma unroll
      for (int j = 0; j < 4; ++j)
#pragma unroll
        for (int e = 0; e < 8; ++e) o[j][e] *= fac[e];

#pragma unroll
      for (int e = 0; e < 8; ++e) { // P (C-layout) -> LDS
        int pr = half * 8 + e;
        pbuf[pr * 40 + lr]      = f2bf(sv[0][e]);
        pbuf[pr * 40 + 16 + lr] = f2bf(sv[1][e]);
      }
      asm volatile("s_wait_dscnt 0" ::: "memory"); // same-wave LDS RAW

      AF pa; // reload P as A-fragment (16x32)
      pa.h[0] = *(const v8bf*)&pbuf[lr * 40 + half * 8];
      pa.h[1] = *(const v8bf*)&pbuf[lr * 40 + 16 + half * 8];
#pragma unroll
      for (int j = 0; j < 4; ++j) {
        AF vb;
        vb.h[0] = *(const v8bf*)&vt[(j * 16 + lr) * 40 + half * 8];
        vb.h[1] = *(const v8bf*)&vt[(j * 16 + lr) * 40 + 16 + half * 8];
        o[j] = __builtin_amdgcn_wmma_f32_16x16x32_bf16(
            false, pa.v, false, vb.v, (short)0, o[j], false, false);
      }
    }
    __syncthreads();
  }

  // epilogue: normalize, write [b][n][h*64+dh] bf16
#pragma unroll
  for (int e = 0; e < 8; ++e) {
    float inv = l_i[e] > 0.f ? 1.f / l_i[e] : 0.f;
    int row = q0 + half * 8 + e;
#pragma unroll
    for (int j = 0; j < 4; ++j)
      Aout[((size_t)(bi * SEQ + row)) * DIMD + hi * 64 + j * 16 + lr] =
          f2bf(o[j][e] * inv);
  }
}

// ---------------- host launcher ----------------
extern "C" void kernel_launch(void* const* d_in, const int* in_sizes, int n_in,
                              void* d_out, int out_size, void* d_ws, size_t ws_size,
                              hipStream_t stream) {
  (void)in_sizes; (void)n_in; (void)out_size; (void)ws_size;
  const float* x    = (const float*)d_in[0];
  const float* g    = (const float*)d_in[1];
  const float* bta  = (const float*)d_in[2];
  const float* wqkv = (const float*)d_in[3];
  const float* wout = (const float*)d_in[4];
  float* out = (float*)d_out;

  const size_t ROWS = (size_t)BATCH * SEQ;          // 8192
  __bf16* xn   = (__bf16*)d_ws;                     // 8192*1024
  __bf16* wq_b = xn + ROWS * DIMD;                  // 1024*3072
  __bf16* wo_b = wq_b + (size_t)DIMD * 3 * DIMD;    // 1024*1024
  __bf16* qkv  = wo_b + (size_t)DIMD * DIMD;        // 3*8192*1024 (q|k|v)[b][h][n][dh]
  __bf16* attn = qkv + 3 * ROWS * DIMD;             // 8192*1024

  ln_kernel<<<(unsigned)ROWS, 256, 0, stream>>>(x, g, bta, xn);
  cast_kernel<<<1024, 256, 0, stream>>>(wqkv, wq_b, DIMD * 3 * DIMD);
  cast_kernel<<<1024, 256, 0, stream>>>(wout, wo_b, DIMD * DIMD);

  gemm_kernel<3 * DIMD, true><<<dim3(24, 64), 256, 0, stream>>>(
      xn, wq_b, qkv, (float*)nullptr, DIMD);

  const __bf16* Qp = qkv;
  const __bf16* Kp = qkv + ROWS * DIMD;
  const __bf16* Vp = qkv + 2 * ROWS * DIMD;
  attn_kernel<<<BATCH * HEADS * (SEQ / 128), 256, 0, stream>>>(Qp, Kp, Vp, attn);

  gemm_kernel<DIMD, false><<<dim3(8, 64), 256, 0, stream>>>(
      attn, wo_b, (__bf16*)nullptr, out, DIMD);
}
